// GIN_74792560493162
// MI455X (gfx1250) — matrix operand order repaired and checked by reading, compiled, not run on hardware
//
#include <hip/hip_runtime.h>
#include <hip/hip_bf16.h>

// ---------------------------------------------------------------------------
// GIN (5 layers) for MI455X / gfx1250: bf16 WMMA GEMMs + f32 atomics scatter.
// GEMM staging uses GLOBAL_LOAD_ASYNC_TO_LDS_B128 (ASYNCcnt) when available,
// with double-buffered LDS tiles overlapping DMA with WMMA.
// ---------------------------------------------------------------------------

typedef __attribute__((ext_vector_type(16))) __bf16 v16bf;
typedef __attribute__((ext_vector_type(8)))  float  v8f;

#define N_NODES 50000
#define N_EDGES 640000
#define IN_DIM  200
#define HID     128
#define NLAYERS 5
#define KPAD0   224   // 200 rounded up to 32

#if defined(__gfx1250__) && __has_builtin(__builtin_amdgcn_global_load_async_to_lds_b128) && __has_builtin(__builtin_amdgcn_s_wait_asynccnt)
#define ASYNC_LDS 1
#else
#define ASYNC_LDS 0
#endif

#if ASYNC_LDS
// builtin signature (from clang diagnostic): param pointee is gcc-style
// vector_size(16) int, in address_space(1) (global) / address_space(3) (LDS).
typedef int v4i_gcc __attribute__((vector_size(16)));
typedef __attribute__((address_space(1))) v4i_gcc* glb_v4i_p;
typedef __attribute__((address_space(3))) v4i_gcc* lds_v4i_p;
#endif

__device__ __forceinline__ unsigned short f2bf(float x) {
    unsigned u = __float_as_uint(x);
    unsigned r = u + 0x7FFFu + ((u >> 16) & 1u);   // round-to-nearest-even
    return (unsigned short)(r >> 16);
}

// stage 32 bytes (16 bf16) global -> LDS
__device__ __forceinline__ void stage32B(const unsigned short* g, unsigned short* l) {
#if ASYNC_LDS
    glb_v4i_p gp = (glb_v4i_p)(uintptr_t)g;
    // LDS flat aperture keeps the byte offset in addr[31:0] (ISA aperture rules)
    lds_v4i_p lp = (lds_v4i_p)(unsigned int)(uintptr_t)l;
    __builtin_amdgcn_global_load_async_to_lds_b128(gp,     lp,     0, 0);
    __builtin_amdgcn_global_load_async_to_lds_b128(gp + 1, lp + 1, 0, 0);
#else
    *(uint4*)l       = *(const uint4*)g;
    *(uint4*)(l + 8) = *(const uint4*)(g + 8);
#endif
}

__device__ __forceinline__ void wait_stage() {
#if ASYNC_LDS
    __builtin_amdgcn_s_wait_asynccnt(0);
#endif
}

// z[i] = (1 + eps) * h[i]
__global__ void init_z_kernel(const float* __restrict__ h,
                              const float* __restrict__ epsp,
                              float* __restrict__ z, size_t n) {
    size_t i = (size_t)blockIdx.x * blockDim.x + threadIdx.x;
    if (i < n) z[i] = (1.0f + *epsp) * h[i];
}

// z[dst[e]] += h[src[e]]  (float4 chunks, f32 atomics)
__global__ void scatter_kernel(const long long* __restrict__ ei,
                               const float* __restrict__ h,
                               float* __restrict__ z,
                               int E, int Kq, int K) {
    size_t i = (size_t)blockIdx.x * blockDim.x + threadIdx.x;
    if (i >= (size_t)E * Kq) return;
    int e = (int)(i / Kq);
    int f = (int)(i % Kq) * 4;
    int s = (int)ei[e];
    int d = (int)ei[(size_t)E + e];
    const float4 v = *(const float4*)(h + (size_t)s * K + f);
    float* o = z + (size_t)d * K + f;
    atomicAdd(o + 0, v.x); atomicAdd(o + 1, v.y);
    atomicAdd(o + 2, v.z); atomicAdd(o + 3, v.w);
}

// Y (bf16, [M][Kpad]) = pad(X f32 [M][K])
__global__ void cvt_bf16_pad_kernel(const float* __restrict__ X,
                                    unsigned short* __restrict__ Y,
                                    int M, int K, int Kpad) {
    size_t i = (size_t)blockIdx.x * blockDim.x + threadIdx.x;
    if (i >= (size_t)M * Kpad) return;
    int k = (int)(i % Kpad);
    size_t m = i / Kpad;
    Y[i] = (k < K) ? f2bf(X[m * (size_t)K + k]) : (unsigned short)0;
}

// Wt (bf16, [128][Kpad]) = transpose(pad(W f32 [K][128]))
__global__ void wcvt_kernel(const float* __restrict__ W,
                            unsigned short* __restrict__ Wt,
                            int K, int Kpad) {
    int i = blockIdx.x * blockDim.x + threadIdx.x;
    if (i >= 128 * Kpad) return;
    int n = i / Kpad, k = i % Kpad;
    Wt[i] = (k < K) ? f2bf(W[(size_t)k * 128 + n]) : (unsigned short)0;
}

__global__ void zero_stats_kernel(float* __restrict__ stats) {
    stats[threadIdx.x] = 0.0f;   // 256 floats: sums[128] + sumsq[128]
}

// per-column sum / sum-of-squares (128 threads = 128 columns)
__global__ void bn_stats_kernel(const float* __restrict__ X,
                                float* __restrict__ stats, int M) {
    int c = threadIdx.x;
    float s = 0.f, q = 0.f;
    for (int r = blockIdx.x; r < M; r += gridDim.x) {
        float v = X[(size_t)r * HID + c];
        s += v; q += v * v;
    }
    atomicAdd(&stats[c], s);
    atomicAdd(&stats[HID + c], q);
}

// BN(train, biased var) + ReLU -> bf16 (feeds next GEMM's A operand)
__global__ void bn_relu_bf16_kernel(const float* __restrict__ X,
                                    const float* __restrict__ stats,
                                    const float* __restrict__ g,
                                    const float* __restrict__ be,
                                    unsigned short* __restrict__ Y,
                                    int M, float invM) {
    size_t i = (size_t)blockIdx.x * blockDim.x + threadIdx.x;
    if (i >= (size_t)M * HID) return;
    int c = (int)(i & (HID - 1));
    float mean = stats[c] * invM;
    float var  = stats[HID + c] * invM - mean * mean;
    float sc   = __frsqrt_rn(var + 1e-5f) * g[c];
    float v    = (X[i] - mean) * sc + be[c];
    Y[i] = f2bf(v > 0.f ? v : 0.f);
}

// BN + ReLU -> f32 (layer output slice in d_out)
__global__ void bn_relu_f32_kernel(const float* __restrict__ X,
                                   const float* __restrict__ stats,
                                   const float* __restrict__ g,
                                   const float* __restrict__ be,
                                   float* __restrict__ Y,
                                   int M, float invM) {
    size_t i = (size_t)blockIdx.x * blockDim.x + threadIdx.x;
    if (i >= (size_t)M * HID) return;
    int c = (int)(i & (HID - 1));
    float mean = stats[c] * invM;
    float var  = stats[HID + c] * invM - mean * mean;
    float sc   = __frsqrt_rn(var + 1e-5f) * g[c];
    float v    = (X[i] - mean) * sc + be[c];
    Y[i] = v > 0.f ? v : 0.f;
}

// ---------------------------------------------------------------------------
// bf16 WMMA GEMM: C[M][128] = A[M][Kpad] * Bt[128][Kpad]^T + bias
// 256 threads (8 wave32). Block tile 128x128, wave tile 16x128 (8 wmma / kstep)
// Double-buffered LDS tiles; async global->LDS staging overlapped with WMMA.
// ---------------------------------------------------------------------------
__global__ __launch_bounds__(256)
void gemm_bf16_kernel(const unsigned short* __restrict__ A,
                      const unsigned short* __restrict__ Bt,
                      const float* __restrict__ bias,
                      float* __restrict__ C,
                      int M, int Kpad) {
    __shared__ __align__(16) unsigned short ldsA[2][128 * 32];
    __shared__ __align__(16) unsigned short ldsB[2][128 * 32];

    const int t    = threadIdx.x;
    const int wave = t >> 5;
    const int lane = t & 31;
    const int lm   = lane & 15;      // fragment row/col within 16
    const int hi   = lane >> 4;      // half-wave select
    const int blockRow = blockIdx.x * 128;

    // Staging assignment: thread -> (tile row rA, 16-element K chunk cA)
    const int rA = t >> 1;           // 0..127 (tile row / weight N)
    const int cA = (t & 1) * 16;     // 0 or 16
    int gr = blockRow + rA;
    if (gr >= M) gr = M - 1;         // clamp: OOB rows feed only unsaved D rows
    const unsigned short* gA = A  + (size_t)gr * Kpad + cA;
    const unsigned short* gB = Bt + (size_t)rA * Kpad + cA;
    const int lo = rA * 32 + cA;     // LDS element offset for this thread

    v8f acc[8];
    #pragma unroll
    for (int i = 0; i < 8; ++i) acc[i] = v8f{};

    // prologue: stage K-slice 0 into buffer 0
    stage32B(gA, &ldsA[0][lo]);
    stage32B(gB, &ldsB[0][lo]);
    wait_stage();
    __syncthreads();

    int buf = 0;
    for (int k0 = 0; k0 < Kpad; k0 += 32) {
        const int nb = buf ^ 1;
        if (k0 + 32 < Kpad) {        // stage next slice while computing this one
            stage32B(gA + k0 + 32, &ldsA[nb][lo]);
            stage32B(gB + k0 + 32, &ldsB[nb][lo]);
        }
        if (k0 + 64 < Kpad)          // gfx1250 global_prefetch_b8 of slice k+2
            __builtin_prefetch(gA + k0 + 64, 0, 0);

        // A fragment: lane(m=lm,hi) VGPR0-3 <- K=8*hi..+7, VGPR4-7 <- K=16+8*hi..+7
        union { uint4 u[2]; v16bf v; } fa;
        const int arow = wave * 16 + lm;
        fa.u[0] = *(const uint4*)&ldsA[buf][arow * 32 + 8 * hi];
        fa.u[1] = *(const uint4*)&ldsA[buf][arow * 32 + 16 + 8 * hi];

        #pragma unroll
        for (int nt = 0; nt < 8; ++nt) {
            // B fragment: lane(n=lm,hi) VGPR0-3 <- K=8*hi..+7, VGPR4-7 <- K=16+8*hi..+7, column n
            union { uint4 u[2]; v16bf v; } fb;
            const int bcol = nt * 16 + lm;
            fb.u[0] = *(const uint4*)&ldsB[buf][bcol * 32 + 8 * hi];
            fb.u[1] = *(const uint4*)&ldsB[buf][bcol * 32 + 16 + 8 * hi];
            acc[nt] = __builtin_amdgcn_wmma_f32_16x16x32_bf16(
                false, fa.v, false, fb.v, (short)0, acc[nt], false, false);
        }

        wait_stage();                // our async writes into nb have landed
        __syncthreads();             // everyone done reading buf / writing nb
        buf = nb;
    }

    // epilogue: D layout VGPR r -> M = r + 8*hi, N = lm
    #pragma unroll
    for (int nt = 0; nt < 8; ++nt) {
        const int col = nt * 16 + lm;
        const float b = bias[col];
        #pragma unroll
        for (int r = 0; r < 8; ++r) {
            const int row = blockRow + wave * 16 + r + 8 * hi;
            if (row < M) C[(size_t)row * HID + col] = acc[nt][r] + b;
        }
    }
}

// ---------------------------------------------------------------------------

extern "C" void kernel_launch(void* const* d_in, const int* in_sizes, int n_in,
                              void* d_out, int out_size, void* d_ws, size_t ws_size,
                              hipStream_t stream) {
    const float*     x  = (const float*)d_in[0];
    const long long* ei = (const long long*)d_in[1];
    // params: d_in[2 + 9*l + {eps,W1,b1,g1,be1,W2,b2,g2,be2}]

    char* ws = (char*)d_ws;
    float*          zbuf = (float*)ws;                                 // N*224 f32 (agg z, reused as t2)
    unsigned short* abf  = (unsigned short*)(ws + (size_t)N_NODES * KPAD0 * 4);      // N*224 bf16 (A operand)
    float*          t1   = (float*)(ws + (size_t)N_NODES * KPAD0 * 4
                                       + (size_t)N_NODES * KPAD0 * 2);               // N*128 f32
    unsigned short* wbf  = (unsigned short*)(ws + (size_t)N_NODES * KPAD0 * 4
                                                + (size_t)N_NODES * KPAD0 * 2
                                                + (size_t)N_NODES * HID * 4);        // bf16 weights

    size_t woff1[NLAYERS], woff2[NLAYERS], woff = 0;
    for (int l = 0; l < NLAYERS; ++l) {
        int kp1 = (l == 0) ? KPAD0 : HID;
        woff1[l] = woff; woff += (size_t)HID * kp1;
        woff2[l] = woff; woff += (size_t)HID * HID;
    }
    float* stats = (float*)(wbf + woff);   // 256 floats

    // ---- convert + transpose + pad all weights to bf16 (cheap, per call) ----
    for (int l = 0; l < NLAYERS; ++l) {
        int K1 = (l == 0) ? IN_DIM : HID;
        int Kp1 = (l == 0) ? KPAD0 : HID;
        const float* W1 = (const float*)d_in[2 + 9 * l + 1];
        const float* W2 = (const float*)d_in[2 + 9 * l + 5];
        wcvt_kernel<<<(128 * Kp1 + 255) / 256, 256, 0, stream>>>(W1, wbf + woff1[l], K1, Kp1);
        wcvt_kernel<<<(128 * HID + 255) / 256, 256, 0, stream>>>(W2, wbf + woff2[l], HID, HID);
    }

    const float invN = 1.0f / (float)N_NODES;
    for (int l = 0; l < NLAYERS; ++l) {
        const float* h   = (l == 0) ? x : ((float*)d_out + (size_t)(l - 1) * N_NODES * HID);
        const int    K   = (l == 0) ? IN_DIM : HID;
        const int    Kp  = (l == 0) ? KPAD0 : HID;
        const float* epsp = (const float*)d_in[2 + 9 * l + 0];
        const float* b1  = (const float*)d_in[2 + 9 * l + 2];
        const float* g1  = (const float*)d_in[2 + 9 * l + 3];
        const float* be1 = (const float*)d_in[2 + 9 * l + 4];
        const float* b2  = (const float*)d_in[2 + 9 * l + 6];
        const float* g2  = (const float*)d_in[2 + 9 * l + 7];
        const float* be2 = (const float*)d_in[2 + 9 * l + 8];

        // GIN aggregation: z = (1+eps)*h + scatter_add(h[src] -> dst)
        size_t nz = (size_t)N_NODES * K;
        init_z_kernel<<<(unsigned)((nz + 255) / 256), 256, 0, stream>>>(h, epsp, zbuf, nz);
        size_t ns = (size_t)N_EDGES * (K / 4);
        scatter_kernel<<<(unsigned)((ns + 255) / 256), 256, 0, stream>>>(ei, h, zbuf, N_EDGES, K / 4, K);

        // GEMM1: t1 = z @ W1 + b1  (bf16 WMMA)
        size_t nc = (size_t)N_NODES * Kp;
        cvt_bf16_pad_kernel<<<(unsigned)((nc + 255) / 256), 256, 0, stream>>>(zbuf, abf, N_NODES, K, Kp);
        gemm_bf16_kernel<<<(N_NODES + 127) / 128, 256, 0, stream>>>(abf, wbf + woff1[l], b1, t1, N_NODES, Kp);

        // BN1 + ReLU -> bf16 A for GEMM2
        zero_stats_kernel<<<1, 256, 0, stream>>>(stats);
        bn_stats_kernel<<<256, 128, 0, stream>>>(t1, stats, N_NODES);
        bn_relu_bf16_kernel<<<(N_NODES * HID + 255) / 256, 256, 0, stream>>>(t1, stats, g1, be1, abf, N_NODES, invN);

        // GEMM2: t2(=zbuf) = a1 @ W2 + b2
        gemm_bf16_kernel<<<(N_NODES + 127) / 128, 256, 0, stream>>>(abf, wbf + woff2[l], b2, zbuf, N_NODES, HID);

        // BN2 + ReLU -> d_out layer slice (f32)
        zero_stats_kernel<<<1, 256, 0, stream>>>(stats);
        bn_stats_kernel<<<256, 128, 0, stream>>>(zbuf, stats, N_NODES);
        bn_relu_f32_kernel<<<(N_NODES * HID + 255) / 256, 256, 0, stream>>>(
            zbuf, stats, g2, be2, (float*)d_out + (size_t)l * N_NODES * HID, N_NODES, invN);
    }
}
